// CausalSelfAttentionGQA_3496103379101
// MI455X (gfx1250) — matrix operand order repaired and checked by reading, not compiled, not run on hardware
//
#include <hip/hip_runtime.h>
#include <hip/hip_bf16.h>
#include <cmath>

// ---------------------------------------------------------------- types
typedef __bf16 bf16;
typedef __attribute__((ext_vector_type(16))) __bf16 v16bf;
typedef __attribute__((ext_vector_type(8)))  __bf16 v8bf;
typedef __attribute__((ext_vector_type(8)))  float  v8f;
typedef __attribute__((ext_vector_type(4)))  float  v4f;
typedef __attribute__((ext_vector_type(4)))  int    v4i;

#define B_    2
#define T_    2048
#define NE_   2048
#define NH_   16
#define NKV_  4
#define DH_   128
#define QKVN_ 3072   // 2048 + 2*4*128

__device__ __forceinline__ v8f zero8() {
  v8f z;
#pragma unroll
  for (int i = 0; i < 8; ++i) z[i] = 0.0f;
  return z;
}
__device__ __forceinline__ v16bf cat8(v8bf lo, v8bf hi) {
  return __builtin_shufflevector(lo, hi, 0,1,2,3,4,5,6,7,8,9,10,11,12,13,14,15);
}

// async direct-to-LDS 16B copy (gfx1250 GLOBAL_LOAD_ASYNC_TO_LDS_B128, ASYNCcnt)
__device__ __forceinline__ void async_ld16(const bf16* g, bf16* l) {
  __builtin_amdgcn_global_load_async_to_lds_b128(
      (__attribute__((address_space(1))) v4i*)g,
      (__attribute__((address_space(3))) v4i*)l, 0, 0);
}
// wait until at most N async ops outstanding (per-wave, in-order completion)
__device__ __forceinline__ void wait_async8() {
  asm volatile("s_wait_asynccnt 0x8" ::: "memory");
}
__device__ __forceinline__ void wait_async0() {
  asm volatile("s_wait_asynccnt 0x0" ::: "memory");
}

// ---------------------------------------------------------------- cast f32 -> bf16 (x8 vectorized)
__global__ void cast_f32_bf16(const float* __restrict__ in, bf16* __restrict__ out, int n8) {
  int i = blockIdx.x * blockDim.x + threadIdx.x;
  if (i < n8) {
    v4f a = *(const v4f*)(in + 8 * (long)i);
    v4f b = *(const v4f*)(in + 8 * (long)i + 4);
    v8bf o;
#pragma unroll
    for (int e = 0; e < 4; ++e) { o[e] = (bf16)a[e]; o[4 + e] = (bf16)b[e]; }
    *(v8bf*)(out + 8 * (long)i) = o;
  }
}

// ---------------------------------------------------------------- WMMA GEMM: C[M,N] = A[M,K]*Bw[K,N]
// block 256 thr (8 waves), tile 128x128, K-step 64, DOUBLE-BUFFERED async-to-LDS.
// wave grid 4(M) x 2(N), each wave 32x64 = 2x4 accumulators, 16 WMMA per stage.
__global__ __launch_bounds__(256)
void gemm_bf16(const bf16* __restrict__ A, const bf16* __restrict__ Bw,
               float* __restrict__ C, int M, int N, int K) {
  __shared__ bf16 At[2][128 * 64];   // 2 x 16 KB
  __shared__ bf16 Bt[2][64 * 128];   // 2 x 16 KB
  const int tid  = threadIdx.x;
  const int lane = tid & 31;
  const int wid  = tid >> 5;
  const int wm   = wid & 3;      // wave row (M)
  const int wn   = wid >> 2;     // wave col (N)
  const int half = lane >> 4;
  const int lr   = lane & 15;
  const long m0  = (long)blockIdx.y * 128;
  const long n0  = (long)blockIdx.x * 128;

  // per-thread staging slots: A: 2 thr/row, 32 elems (64B) each; B: 4 thr/row
  const int ar = tid >> 1, ac = (tid & 1) * 32;
  const int br = tid >> 2, bc = (tid & 3) * 32;
  const bf16* gA = A + (m0 + ar) * K + ac;
  const bf16* gB = Bw + (long)br * N + n0 + bc;
  bf16* lA[2] = { &At[0][ar * 64 + ac],  &At[1][ar * 64 + ac]  };
  bf16* lB[2] = { &Bt[0][br * 128 + bc], &Bt[1][br * 128 + bc] };

  // stage one 64-wide K tile (8 async b128 per thread, A/B interleaved)
  auto stage = [&](int k0, int buf) {
#pragma unroll
    for (int q = 0; q < 4; ++q) {
      async_ld16(gA + k0 + q * 8, lA[buf] + q * 8);
      async_ld16(gB + (long)k0 * N + q * 8, lB[buf] + q * 8);
    }
  };

  v8f acc[2][4];
#pragma unroll
  for (int i = 0; i < 2; ++i)
#pragma unroll
    for (int j = 0; j < 4; ++j) acc[i][j] = zero8();

  stage(0, 0);
  int cur = 0;
  for (int k0 = 0; k0 < K; k0 += 64) {
    if (k0 + 64 < K) {
      // prefetch-issue next tile into the other buffer, overlap with compute
      stage(k0 + 64, cur ^ 1);
      wait_async8();           // oldest 8 (current tile) complete
    } else {
      wait_async0();           // tail: drain
    }
    __syncthreads();

#pragma unroll
    for (int ks = 0; ks < 2; ++ks) {
      v16bf af[2];
#pragma unroll
      for (int i = 0; i < 2; ++i) {
        int m = wm * 32 + i * 16 + lr;
        v8bf lo = *(const v8bf*)&At[cur][m * 64 + ks * 32 + half * 8];
        v8bf hi = *(const v8bf*)&At[cur][m * 64 + ks * 32 + 16 + half * 8];
        af[i] = cat8(lo, hi);
      }
      v16bf bf_[4];
#pragma unroll
      for (int j = 0; j < 4; ++j)
#pragma unroll
        for (int s = 0; s < 16; ++s)
          bf_[j][s] = Bt[cur][(ks * 32 + (s < 8 ? half * 8 + s : 8 + half * 8 + s)) * 128 + wn * 64 + j * 16 + lr];

#pragma unroll
      for (int i = 0; i < 2; ++i)
#pragma unroll
        for (int j = 0; j < 4; ++j)
          acc[i][j] = __builtin_amdgcn_wmma_f32_16x16x32_bf16(
              false, af[i], false, bf_[j], (short)0, acc[i][j], false, false);
    }
    __syncthreads();   // all waves done with 'cur' -> safe to overwrite next iter
    cur ^= 1;
  }

  // epilogue: one 64-bit base per output row, constant-offset stores
#pragma unroll
  for (int i = 0; i < 2; ++i)
#pragma unroll
    for (int r = 0; r < 8; ++r) {
      float* rp = C + (m0 + wm * 32 + i * 16 + r + 8 * half) * (long)N + n0 + wn * 64 + lr;
#pragma unroll
      for (int j = 0; j < 4; ++j) rp[j * 16] = acc[i][j][r];
    }
}

// ---------------------------------------------------------------- YaRN RoPE
struct YarnArgs { float inv_freq[64]; float attn_q; float attn_k; };

// one thread per (b,t,head 0..19,pair 0..63). heads 0..15 = Q, 16..19 = K.
__global__ void rope_kernel(const float* __restrict__ qkv, bf16* __restrict__ qb,
                            bf16* __restrict__ ktb, YarnArgs ya) {
  long idx = (long)blockIdx.x * blockDim.x + threadIdx.x;
  int  i   = idx & 63;
  int  h   = (int)((idx >> 6) % 20);
  long bt  = idx / (64 * 20);
  int  t   = (int)(bt % T_);
  int  b   = (int)(bt / T_);
  float ang = (float)t * ya.inv_freq[i];
  float c = cosf(ang), s = sinf(ang);
  if (h < NH_) {
    long base = (bt)*QKVN_ + (long)h * DH_;
    float e = qkv[base + 2 * i], o = qkv[base + 2 * i + 1];
    long ob = (((long)(b * NH_ + h)) * T_ + t) * DH_;
    qb[ob + 2 * i]     = (bf16)((e * c - o * s) * ya.attn_q);
    qb[ob + 2 * i + 1] = (bf16)((e * s + o * c) * ya.attn_q);
  } else {
    int kh = h - NH_;
    long base = (bt)*QKVN_ + NE_ + (long)kh * DH_;
    float e = qkv[base + 2 * i], o = qkv[base + 2 * i + 1];
    long ob = (long)(b * NKV_ + kh) * DH_ * T_;     // [b][kv][d][t]
    ktb[ob + (long)(2 * i) * T_ + t]     = (bf16)((e * c - o * s) * ya.attn_k);
    ktb[ob + (long)(2 * i + 1) * T_ + t] = (bf16)((e * s + o * c) * ya.attn_k);
  }
}

// V pass-through: [b][kv][t][d] bf16
__global__ void pack_v(const float* __restrict__ qkv, bf16* __restrict__ vb) {
  long idx = (long)blockIdx.x * blockDim.x + threadIdx.x;  // B*T*512
  int  j   = idx & 511;          // kh*128 + d
  long bt  = idx >> 9;
  int  t   = (int)(bt % T_);
  int  b   = (int)(bt / T_);
  int  kh  = j >> 7, d = j & 127;
  vb[(((long)(b * NKV_ + kh)) * T_ + t) * DH_ + d] =
      (bf16)qkv[bt * QKVN_ + NE_ + NKV_ * DH_ + j];
}

// ---------------------------------------------------------------- flash attention (1 wave / 16-row q tile)
__global__ __launch_bounds__(32)
void attn_kernel(const bf16* __restrict__ qb, const bf16* __restrict__ ktb,
                 const bf16* __restrict__ vb, bf16* __restrict__ yb) {
  __shared__ bf16 Pl[16 * 32];
  const int qt  = blockIdx.x;        // 0..127
  const int h   = blockIdx.y;        // 0..15
  const int b   = blockIdx.z;        // 0..1
  const int kvh = h >> 2;            // G = 4
  const int lane = threadIdx.x, half = lane >> 4, lr = lane & 15;

  const bf16* Q  = qb  + ((long)(b * NH_ + h)) * T_ * DH_;
  const bf16* Kt = ktb + ((long)(b * NKV_ + kvh)) * DH_ * T_;
  const bf16* V  = vb  + ((long)(b * NKV_ + kvh)) * T_ * DH_;

  // Q fragments (16 x 128 as 4 A-fragments)
  v16bf qf[4];
  const int trow = qt * 16 + lr;
#pragma unroll
  for (int c = 0; c < 4; ++c) {
    v8bf lo = *(const v8bf*)&Q[trow * DH_ + c * 32 + half * 8];
    v8bf hi = *(const v8bf*)&Q[trow * DH_ + c * 32 + 16 + half * 8];
    qf[c] = cat8(lo, hi);
  }

  v8f o[8];
#pragma unroll
  for (int j = 0; j < 8; ++j) o[j] = zero8();
  float mrow[8], lsum[8];
#pragma unroll
  for (int r = 0; r < 8; ++r) { mrow[r] = -1e30f; lsum[r] = 0.0f; }

  const int nkb = (qt * 16 + 47) / 32;   // 32-key blocks covering causal range
  for (int kb = 0; kb < nkb; ++kb) {
    const int k0 = kb * 32;
    // S = Q * K^T  (two 16x16 tiles over 32 keys)
    v8f s0 = zero8(), s1 = zero8();
#pragma unroll
    for (int c = 0; c < 4; ++c) {
      v16bf kf0, kf1;
#pragma unroll
      for (int s = 0; s < 16; ++s) {
        const int kk = c * 32 + (s < 8 ? half * 8 + s : 8 + half * 8 + s);
        kf0[s] = Kt[(long)kk * T_ + k0 + lr];
        kf1[s] = Kt[(long)kk * T_ + k0 + 16 + lr];
      }
      s0 = __builtin_amdgcn_wmma_f32_16x16x32_bf16(false, qf[c], false, kf0, (short)0, s0, false, false);
      s1 = __builtin_amdgcn_wmma_f32_16x16x32_bf16(false, qf[c], false, kf1, (short)0, s1, false, false);
    }

    // causal mask + online softmax (row = r + 8*half, cols = k0+lr / k0+16+lr)
    float alpha[8];
#pragma unroll
    for (int r = 0; r < 8; ++r) {
      const int rt = qt * 16 + r + 8 * half;
      float a  = (k0 + lr      > rt) ? -1e30f : s0[r];
      float bb = (k0 + 16 + lr > rt) ? -1e30f : s1[r];
      float mx = fmaxf(a, bb);
      mx = fmaxf(mx, __shfl_xor(mx, 1, 32));
      mx = fmaxf(mx, __shfl_xor(mx, 2, 32));
      mx = fmaxf(mx, __shfl_xor(mx, 4, 32));
      mx = fmaxf(mx, __shfl_xor(mx, 8, 32));
      const float mn = fmaxf(mrow[r], mx);
      const float al = __expf(mrow[r] - mn);
      const float e0 = __expf(a - mn);
      const float e1 = __expf(bb - mn);
      float sm = e0 + e1;
      sm += __shfl_xor(sm, 1, 32);
      sm += __shfl_xor(sm, 2, 32);
      sm += __shfl_xor(sm, 4, 32);
      sm += __shfl_xor(sm, 8, 32);
      lsum[r] = lsum[r] * al + sm;
      mrow[r] = mn;
      alpha[r] = al;
      Pl[(r + 8 * half) * 32 + lr]      = (bf16)e0;
      Pl[(r + 8 * half) * 32 + 16 + lr] = (bf16)e1;
    }
    __syncthreads();   // single-wave WG: orders LDS write -> read

    v16bf pf;
    {
      v8bf lo = *(const v8bf*)&Pl[lr * 32 + half * 8];
      v8bf hi = *(const v8bf*)&Pl[lr * 32 + 16 + half * 8];
      pf = cat8(lo, hi);
    }
    __syncthreads();   // protect Pl from next iteration's writes

    // O = alpha*O + P*V
#pragma unroll
    for (int j = 0; j < 8; ++j) {
      v16bf vf;
#pragma unroll
      for (int s = 0; s < 16; ++s)
        vf[s] = V[(long)(k0 + (s < 8 ? half * 8 + s : 8 + half * 8 + s)) * DH_ + j * 16 + lr];
#pragma unroll
      for (int r = 0; r < 8; ++r) o[j][r] *= alpha[r];
      o[j] = __builtin_amdgcn_wmma_f32_16x16x32_bf16(false, pf, false, vf, (short)0, o[j], false, false);
    }
  }

  // normalize + write y (bf16, [b*T + t][2048])
#pragma unroll
  for (int r = 0; r < 8; ++r) lsum[r] = 1.0f / lsum[r];
#pragma unroll
  for (int j = 0; j < 8; ++j)
#pragma unroll
    for (int r = 0; r < 8; ++r) {
      long row = (long)b * T_ + qt * 16 + r + 8 * half;
      long col = (long)h * DH_ + j * 16 + lr;
      yb[row * NE_ + col] = (bf16)(o[j][r] * lsum[r]);
    }
}

// ---------------------------------------------------------------- launch
extern "C" void kernel_launch(void* const* d_in, const int* in_sizes, int n_in,
                              void* d_out, int out_size, void* d_ws, size_t ws_size,
                              hipStream_t stream) {
  const float* x     = (const float*)d_in[0];
  const float* w_qkv = (const float*)d_in[1];
  const float* w_o   = (const float*)d_in[2];
  float* out = (float*)d_out;

  // workspace carve-up (element counts)
  const long XB_E = (long)B_ * T_ * NE_;        // 8,388,608
  const long WQ_E = (long)NE_ * QKVN_;          // 6,291,456
  const long WO_E = (long)NE_ * NE_;            // 4,194,304
  const long QKV_E = (long)B_ * T_ * QKVN_;     // 12,582,912 (f32)
  const long QB_E = (long)B_ * NH_ * T_ * DH_;  // 8,388,608
  const long KT_E = (long)B_ * NKV_ * DH_ * T_; // 2,097,152
  const long VB_E = KT_E;

  bf16* xb    = (bf16*)d_ws;
  bf16* wqkvb = xb + XB_E;
  bf16* wob   = wqkvb + WQ_E;
  float* qkv  = (float*)(wob + WO_E);
  bf16* qb    = (bf16*)(qkv + QKV_E);
  bf16* ktb   = qb + QB_E;
  bf16* vb    = ktb + KT_E;
  bf16* yb    = vb + VB_E;

  // host-side YaRN constants (double, matching numpy)
  YarnArgs ya;
  {
    const double base = 1000000.0, factor = 64.0;
    const int dim = DH_;
    auto corr = [&](double nr) {
      return dim * log(4096.0 / (nr * 2.0 * M_PI)) / (2.0 * log(base));
    };
    double low  = floor(corr(4.0)); if (low < 0.0) low = 0.0;
    double high = ceil(corr(1.0));  if (high > dim - 1) high = dim - 1;
    if (low == high) high += 0.001;
    for (int i = 0; i < 64; ++i) {
      double t = (double)i;
      double ramp = (t - low) / (high - low);
      ramp = ramp < 0.0 ? 0.0 : (ramp > 1.0 ? 1.0 : ramp);
      double pf = pow(base, (2.0 * i) / dim);
      double inv_extrap = 1.0 / pf;
      double inv_interp = 1.0 / (factor * pf);
      double extrap = 1.0 - ramp;
      ya.inv_freq[i] = (float)(inv_interp * (1.0 - extrap) + inv_extrap * extrap);
    }
    double attn_factor = 0.1 * log(factor) + 1.0;
    ya.attn_k = (float)attn_factor;
    ya.attn_q = (float)(attn_factor / sqrt((double)DH_));
  }

  // 1) casts
  cast_f32_bf16<<<(int)(XB_E / 8 / 256), 256, 0, stream>>>(x, xb, (int)(XB_E / 8));
  cast_f32_bf16<<<(int)(WQ_E / 8 / 256), 256, 0, stream>>>(w_qkv, wqkvb, (int)(WQ_E / 8));
  cast_f32_bf16<<<(int)(WO_E / 8 / 256), 256, 0, stream>>>(w_o, wob, (int)(WO_E / 8));

  // 2) qkv = x @ w_qkv   (4096 x 3072, K=2048)
  gemm_bf16<<<dim3(QKVN_ / 128, (B_ * T_) / 128), 256, 0, stream>>>(
      xb, wqkvb, qkv, B_ * T_, QKVN_, NE_);

  // 3) RoPE + layouts
  {
    long n = (long)B_ * T_ * 20 * 64;
    rope_kernel<<<(int)(n / 256), 256, 0, stream>>>(qkv, qb, ktb, ya);
  }
  {
    long n = (long)B_ * T_ * NKV_ * DH_;
    pack_v<<<(int)(n / 256), 256, 0, stream>>>(qkv, vb);
  }

  // 4) flash attention
  attn_kernel<<<dim3(T_ / 16, NH_, B_), 32, 0, stream>>>(qb, ktb, vb, yb);

  // 5) out = y @ w_o   (4096 x 2048, K=2048)
  gemm_bf16<<<dim3(NE_ / 128, (B_ * T_) / 128), 256, 0, stream>>>(
      yb, wob, out, B_ * T_, NE_, NE_);
}
